// UVNetHeteroGraphEncoder_83227876261955
// MI455X (gfx1250) — matrix operand order repaired and checked by reading, not compiled, not run on hardware
//
#include <hip/hip_runtime.h>
#include <hip/hip_bf16.h>

// ---------------------------------------------------------------------------
// UVNet hetero-graph encoder for gfx1250 (MI455X), wave32 + WMMA bf16.
//
// Heavy math: per-edge NNConv weights  W = relu(ef@A)@Bm  reformulated as
//   msg = (R (x) hsrc) @ Bm2   with Bm2[j*32+i, o] = Bm[j, i*32+o]
// so everything is bf16 WMMA with f32 accumulate and the 512MB/etype
// intermediate W is never materialized. All weights are pre-shuffled into
// WMMA B-fragment order so every B operand is two global_load_b128.
// ---------------------------------------------------------------------------

#define NA_  16384
#define NB_  8192
#define BGR_ 8
#define E1_  131072
#define E2_  65536

typedef __attribute__((ext_vector_type(16))) __bf16 bf16x16;
typedef __attribute__((ext_vector_type(8)))  float  f32x8;

// ---- WMMA helpers ----------------------------------------------------------
// A-matrix 16x32 bf16 layout (ISA 7.12.2): lane L holds row M=L&15, half=L>>4;
// element v of the v16 register file maps to K = ((v&8)<<1) + half*8 + (v&7).
__device__ __forceinline__ int kmap(int v, int half) {
  return ((v & 8) << 1) + half * 8 + (v & 7);
}

__device__ __forceinline__ f32x8 wmma_bf16(bf16x16 a, bf16x16 b, f32x8 c) {
  // (neg_a, A, neg_b, B, c_mod, C, reuse_a, reuse_b)
  return __builtin_amdgcn_wmma_f32_16x16x32_bf16(false, a, false, b, (short)0, c,
                                                 false, false);
}

// Build A fragment from an LDS f32 tile [16][stride], K chunk at kb.
__device__ __forceinline__ bf16x16 make_a_frag(const float* tile, int stride,
                                               int kb, int lane) {
  int m = lane & 15, half = lane >> 4;
  bf16x16 a;
#pragma unroll
  for (int v = 0; v < 16; ++v)
    a[v] = (__bf16)tile[m * stride + kb + kmap(v, half)];
  return a;
}

// Load a pre-shuffled B fragment: packed layout [kc][nt][lane][16].
__device__ __forceinline__ bf16x16 load_b_frag(const __bf16* w, int kc, int nt,
                                               int lane) {
  return ((const bf16x16*)w)[(kc * 2 + nt) * 32 + lane];
}

// ---- weight packing --------------------------------------------------------
// Pack W [Din<=Kpad][32] f32 into bf16 B-fragment order:
// out[((kc*2+nt)*32+lane)*16 + v] = W[k, n], k = kc*32 + ((v&8)<<1) + (lane>>4)*8 + (v&7),
// n = nt*16 + (lane&15); zero-pad k >= Din.
__global__ void pack_bfrag_kernel(const float* __restrict__ W,
                                  __bf16* __restrict__ out, int Din, int Kpad) {
  int id = blockIdx.x * blockDim.x + threadIdx.x;
  if (id >= (Kpad >> 5) * 1024) return;
  int v = id & 15, lane = (id >> 4) & 31, nt = (id >> 9) & 1, kc = id >> 10;
  int k = kc * 32 + ((v & 8) << 1) + (lane >> 4) * 8 + (v & 7);
  int n = nt * 16 + (lane & 15);
  out[id] = (k < Din) ? (__bf16)W[k * 32 + n] : (__bf16)0.0f;
}

// Bm [32][1024] -> Bm2 [1024][32] in B-fragment order (layout [nt][kc][lane][16]):
// k = kc*32 + ((v&8)<<1) + (lane>>4)*8 + (v&7), n = nt*16 + (lane&15),
// Bm2[k,n] = Bm[k>>5, (k&31)*32 + n].
__global__ void pack_bm2_kernel(const float* __restrict__ Bm,
                                __bf16* __restrict__ out) {
  int id = blockIdx.x * blockDim.x + threadIdx.x;
  if (id >= 32768) return;
  int v = id & 15, lane = (id >> 4) & 31, kc = (id >> 9) & 31, nt = id >> 14;
  int k = kc * 32 + ((v & 8) << 1) + (lane >> 4) * 8 + (v & 7);
  int n = nt * 16 + (lane & 15);
  int j = k >> 5, i = k & 31;
  out[id] = (__bf16)Bm[j * 1024 + i * 32 + n];
}

// ---- generic MLP2 + LayerNorm:  Y = LN(W2 @ lrelu(W1 @ (s*X (+Add)) + b1) + b2)
// 128 threads = 4 waves, each wave owns one 16-row tile. grid = M/64.
__global__ __launch_bounds__(128) void mlp_ln_kernel(
    const float* __restrict__ X, const float* __restrict__ Add,
    const float* __restrict__ eps,
    const __bf16* __restrict__ W1, const float* __restrict__ b1,
    const __bf16* __restrict__ W2, const float* __restrict__ b2,
    const float* __restrict__ g, const float* __restrict__ beta,
    float* __restrict__ Y, int Din, int Kpad) {
  __shared__ float xs[4][16][64];
  __shared__ float hs[4][16][32];
  int wave = threadIdx.x >> 5, lane = threadIdx.x & 31;
  int half = lane >> 4, nlo = lane & 15;
  int row0 = (blockIdx.x * 4 + wave) * 16;
  float scale = 1.0f + (eps ? *eps : 0.0f);

  // vectorized tile load: Din,Kpad are multiples of 16 so float4 never
  // straddles the Din boundary.
  int kp4 = Kpad >> 2;
  for (int e = lane; e < 16 * kp4; e += 32) {
    int r = e / kp4, c4 = e - r * kp4, c = c4 * 4;
    float4 v = make_float4(0.f, 0.f, 0.f, 0.f);
    if (c < Din) {
      float4 x = *(const float4*)(X + (size_t)(row0 + r) * Din + c);
      v = make_float4(scale * x.x, scale * x.y, scale * x.z, scale * x.w);
    }
    if (Add && c < 32) {
      float4 a = *(const float4*)(Add + (size_t)(row0 + r) * 32 + c);
      v.x += a.x; v.y += a.y; v.z += a.z; v.w += a.w;
    }
    *(float4*)&xs[wave][r][c] = v;
  }
  __syncthreads();

  f32x8 c0 = {}, c1 = {};
  for (int kb = 0; kb < Kpad; kb += 32) {
    int kc = kb >> 5;
    bf16x16 af = make_a_frag(&xs[wave][0][0], 64, kb, lane);
    c0 = wmma_bf16(af, load_b_frag(W1, kc, 0, lane), c0);
    c1 = wmma_bf16(af, load_b_frag(W1, kc, 1, lane), c1);
  }
  float bb0 = b1[nlo], bb1 = b1[16 + nlo];
#pragma unroll
  for (int r = 0; r < 8; ++r) {
    float v0 = c0[r] + bb0; v0 = v0 > 0.f ? v0 : 0.01f * v0;  // leaky_relu(0.01)
    float v1 = c1[r] + bb1; v1 = v1 > 0.f ? v1 : 0.01f * v1;
    hs[wave][half * 8 + r][nlo] = v0;
    hs[wave][half * 8 + r][16 + nlo] = v1;
  }
  __syncthreads();

  f32x8 d0 = {}, d1 = {};
  {
    bf16x16 af = make_a_frag(&hs[wave][0][0], 32, 0, lane);
    d0 = wmma_bf16(af, load_b_frag(W2, 0, 0, lane), d0);
    d1 = wmma_bf16(af, load_b_frag(W2, 0, 1, lane), d1);
  }
  float cb0 = b2[nlo], cb1 = b2[16 + nlo];
#pragma unroll
  for (int r = 0; r < 8; ++r) {
    xs[wave][half * 8 + r][nlo] = d0[r] + cb0;
    xs[wave][half * 8 + r][16 + nlo] = d1[r] + cb1;
  }
  __syncthreads();

  if (lane < 16) {  // LayerNorm, one row per lane
    float mu = 0.f;
#pragma unroll
    for (int j = 0; j < 32; ++j) mu += xs[wave][lane][j];
    mu *= (1.0f / 32.0f);
    float var = 0.f;
#pragma unroll
    for (int j = 0; j < 32; ++j) { float d = xs[wave][lane][j] - mu; var += d * d; }
    var *= (1.0f / 32.0f);
    float inv = rsqrtf(var + 1e-5f);
#pragma unroll
    for (int j4 = 0; j4 < 8; ++j4) {
      float4 o;
      o.x = g[j4 * 4 + 0] * (xs[wave][lane][j4 * 4 + 0] - mu) * inv + beta[j4 * 4 + 0];
      o.y = g[j4 * 4 + 1] * (xs[wave][lane][j4 * 4 + 1] - mu) * inv + beta[j4 * 4 + 1];
      o.z = g[j4 * 4 + 2] * (xs[wave][lane][j4 * 4 + 2] - mu) * inv + beta[j4 * 4 + 2];
      o.w = g[j4 * 4 + 3] * (xs[wave][lane][j4 * 4 + 3] - mu) * inv + beta[j4 * 4 + 3];
      *(float4*)(Y + (size_t)(row0 + lane) * 32 + j4 * 4) = o;
    }
  }
}

// ---- fused NNConv: R = relu(ef@A); msg = (R (x) (1+eps)h[src]) @ Bm2;
//      conv[dst] += msg (atomic). 128 threads / block, each wave does 4 tiles.
__global__ __launch_bounds__(128) void nnconv_kernel(
    const float* __restrict__ ef, const float* __restrict__ hn,
    const int* __restrict__ src, const int* __restrict__ dst,
    const float* __restrict__ epsn,
    const __bf16* __restrict__ Abf, const __bf16* __restrict__ Bm2f,
    float* __restrict__ conv) {
  __shared__ float rt[4][4][16][32];  // per-wave R tiles
  __shared__ float efs[4][16][32];    // per-wave ef staging
  int wave = threadIdx.x >> 5, lane = threadIdx.x & 31;
  int half = lane >> 4, m = lane & 15;
  float scale = 1.0f + *epsn;
  int base = blockIdx.x * 256 + wave * 64;

  __builtin_prefetch(&ef[(size_t)base * 32], 0, 1);  // global_prefetch_b8

  // stage: R = relu(ef @ A) per tile, via WMMA
  for (int t = 0; t < 4; ++t) {
    int tb = base + t * 16;
    {  // b128 staging: lane -> (row = lane>>1, 16-col half)
      int r = lane >> 1, c0i = (lane & 1) * 16;
      const float4* s4 = (const float4*)(ef + (size_t)(tb + r) * 32 + c0i);
      float4* d4 = (float4*)&efs[wave][r][c0i];
      d4[0] = s4[0]; d4[1] = s4[1]; d4[2] = s4[2]; d4[3] = s4[3];
    }
    __syncthreads();
    f32x8 c0 = {}, c1 = {};
    bf16x16 af = make_a_frag(&efs[wave][0][0], 32, 0, lane);
    c0 = wmma_bf16(af, load_b_frag(Abf, 0, 0, lane), c0);
    c1 = wmma_bf16(af, load_b_frag(Abf, 0, 1, lane), c1);
#pragma unroll
    for (int r = 0; r < 8; ++r) {
      rt[wave][t][half * 8 + r][m] = fmaxf(c0[r], 0.0f);
      rt[wave][t][half * 8 + r][16 + m] = fmaxf(c1[r], 0.0f);
    }
    __syncthreads();
  }

  // gather hsrc values matching the A-fragment K layout: per tile this is
  // four contiguous float4 runs per lane (b128 loads).
  float hsv[4][16];
#pragma unroll
  for (int t = 0; t < 4; ++t) {
    int s = src[base + t * 16 + m];
    const float* row = hn + (size_t)s * 32;
    float4 a0 = *(const float4*)(row + half * 8);
    float4 a1 = *(const float4*)(row + half * 8 + 4);
    float4 b0 = *(const float4*)(row + 16 + half * 8);
    float4 b1 = *(const float4*)(row + 16 + half * 8 + 4);
    hsv[t][0] = scale * a0.x;  hsv[t][1] = scale * a0.y;
    hsv[t][2] = scale * a0.z;  hsv[t][3] = scale * a0.w;
    hsv[t][4] = scale * a1.x;  hsv[t][5] = scale * a1.y;
    hsv[t][6] = scale * a1.z;  hsv[t][7] = scale * a1.w;
    hsv[t][8] = scale * b0.x;  hsv[t][9] = scale * b0.y;
    hsv[t][10] = scale * b0.z; hsv[t][11] = scale * b0.w;
    hsv[t][12] = scale * b1.x; hsv[t][13] = scale * b1.y;
    hsv[t][14] = scale * b1.z; hsv[t][15] = scale * b1.w;
  }

  f32x8 zero = {};
  f32x8 acc[4][2];
#pragma unroll
  for (int t = 0; t < 4; ++t) { acc[t][0] = zero; acc[t][1] = zero; }

  const bf16x16* B = (const bf16x16*)Bm2f;  // frag-packed: [nt][kc][lane]
  for (int kc = 0; kc < 32; ++kc) {
    bf16x16 b0 = B[(0 * 32 + kc) * 32 + lane];
    bf16x16 b1 = B[(1 * 32 + kc) * 32 + lane];
#pragma unroll
    for (int t = 0; t < 4; ++t) {
      float rv = rt[wave][t][m][kc];  // j == kc within this K chunk
      bf16x16 af;
#pragma unroll
      for (int v = 0; v < 16; ++v) af[v] = (__bf16)(rv * hsv[t][v]);
      acc[t][0] = wmma_bf16(af, b0, acc[t][0]);
      acc[t][1] = wmma_bf16(af, b1, acc[t][1]);
    }
  }

  // scatter-add messages
#pragma unroll
  for (int t = 0; t < 4; ++t) {
    int dd[8];
#pragma unroll
    for (int r = 0; r < 8; ++r) dd[r] = dst[base + t * 16 + half * 8 + r];
#pragma unroll
    for (int r = 0; r < 8; ++r) {
      atomicAdd(&conv[(size_t)dd[r] * 32 + m], acc[t][0][r]);
      atomicAdd(&conv[(size_t)dd[r] * 32 + 16 + m], acc[t][1][r]);
    }
  }
}

// ---- pooling ---------------------------------------------------------------
__global__ void pool_reduce_kernel(const float* __restrict__ h,
                                   const int* __restrict__ gid, int N,
                                   float* __restrict__ msum,
                                   float* __restrict__ cnt) {
  int i = blockIdx.x * blockDim.x + threadIdx.x;
  if (i >= N * 32) return;
  int node = i >> 5, j = i & 31;
  int gidx = gid[node];
  atomicAdd(&msum[gidx * 32 + j], h[(size_t)node * 32 + j]);
  if (j == 0) atomicAdd(&cnt[gidx], 1.0f);
}

__global__ __launch_bounds__(512) void pool_out_kernel(
    const float* __restrict__ mA, const float* __restrict__ mB,
    const float* __restrict__ cA, const float* __restrict__ cB,
    const float* __restrict__ P1, const float* __restrict__ pb1,
    const float* __restrict__ P2, const float* __restrict__ pb2,
    const float* __restrict__ Wo, const float* __restrict__ bo,
    float* __restrict__ out) {
  __shared__ float z[BGR_][2][32];
  __shared__ float w[BGR_][2];
  int t = threadIdx.x;
  {
    int b = t >> 6, ty = (t >> 5) & 1, d = t & 31;
    const float* ms = ty ? mB : mA;
    const float* cs = ty ? cB : cA;
    z[b][ty][d] = ms[b * 32 + d] / cs[b];
  }
  __syncthreads();
  if (t < 16) {
    int b = t >> 1, ty = t & 1;
    float acc = pb2[0];
    for (int hh = 0; hh < 128; ++hh) {
      float s = pb1[hh];
      for (int d = 0; d < 32; ++d) s += z[b][ty][d] * P1[d * 128 + hh];
      acc += tanhf(s) * P2[hh];
    }
    w[b][ty] = acc;
  }
  __syncthreads();
  {
    int b = t >> 6, o = t & 63;
    float w0 = w[b][0], w1 = w[b][1];
    float mx = fmaxf(w0, w1);
    float e0 = expf(w0 - mx), e1 = expf(w1 - mx);
    float inv = 1.0f / (e0 + e1);
    float be0 = e0 * inv, be1 = e1 * inv;
    float acc = bo[o];
    for (int d = 0; d < 32; ++d)
      acc += (be0 * z[b][0][d] + be1 * z[b][1][d]) * Wo[d * 64 + o];
    out[b * 64 + o] = acc;
  }
}

// ---------------------------------------------------------------------------
extern "C" void kernel_launch(void* const* d_in, const int* in_sizes, int n_in,
                              void* d_out, int out_size, void* d_ws, size_t ws_size,
                              hipStream_t stream) {
  (void)in_sizes; (void)n_in; (void)out_size; (void)ws_size;

  const float* xA = (const float*)d_in[0];
  const float* xB = (const float*)d_in[1];
  const float* eA = (const float*)d_in[2];
  const float* eB = (const float*)d_in[3];
  const int* src1 = (const int*)d_in[4];
  const int* dst1 = (const int*)d_in[5];
  const int* src2 = (const int*)d_in[6];
  const int* dst2 = (const int*)d_in[7];
  const int* gidA = (const int*)d_in[8];
  const int* gidB = (const int*)d_in[9];

  // params flattened in JAX pytree (sorted-key) order starting at d_in[10]:
  // per layer (24 leaves): e1.{W1,W2,b1,b2,beta,g}, e2.{W1,W2,b1,b2,beta,g},
  //   eps_e, node.{A.et1,A.et2,Bm.et1,Bm.et2,W1,W2,b1,b2,beta,eps_n,g}
  // then out.{Wo,bo}, pool.{P1,P2,pb1,pb2}
  auto PRM = [&](int i) { return (const float*)d_in[10 + i]; };

  // --- workspace carve-out (~30 MB) ---
  char* wsb = (char*)d_ws;
  size_t off = 0;
  auto alloc = [&](size_t bytes) -> void* {
    void* p = wsb + off;
    off = (off + bytes + 255) & ~(size_t)255;
    return p;
  };
  float* ef1 = (float*)alloc((size_t)E1_ * 32 * 4);
  float* ef2 = (float*)alloc((size_t)E2_ * 32 * 4);
  float* hA = (float*)alloc((size_t)NA_ * 32 * 4);
  float* hB = (float*)alloc((size_t)NB_ * 32 * 4);
  float* conv = (float*)alloc((size_t)NA_ * 32 * 4);
  float* mA = (float*)alloc(BGR_ * 32 * 4);
  float* mB = (float*)alloc(BGR_ * 32 * 4);
  float* cA = (float*)alloc(BGR_ * 4);
  float* cB = (float*)alloc(BGR_ * 4);
  __bf16* pw[2][8];
  __bf16* pbm[2][2];
  for (int l = 0; l < 2; ++l) {
    for (int s = 0; s < 8; ++s) pw[l][s] = (__bf16*)alloc(2048 * 2);
    pbm[l][0] = (__bf16*)alloc(32768 * 2);
    pbm[l][1] = (__bf16*)alloc(32768 * 2);
  }

  // --- pack weights to bf16 fragment order (tiny; every call, deterministic)
  auto packw = [&](const float* W, __bf16* out, int Din, int Kpad) {
    int n = (Kpad >> 5) * 1024;
    pack_bfrag_kernel<<<(n + 255) / 256, 256, 0, stream>>>(W, out, Din, Kpad);
  };
  for (int l = 0; l < 2; ++l) {
    int lb = l * 24;
    int kp1 = l ? 32 : 64;  // et1 edge W1 Din/Kpad
    int di2 = l ? 32 : 16;  // et2 edge W1 Din (padded to 32)
    packw(PRM(lb + 0), pw[l][0], kp1, kp1);
    packw(PRM(lb + 1), pw[l][1], 32, 32);
    packw(PRM(lb + 6), pw[l][2], di2, 32);
    packw(PRM(lb + 7), pw[l][3], 32, 32);
    packw(PRM(lb + 13), pw[l][4], 32, 32);
    packw(PRM(lb + 14), pw[l][5], 32, 32);
    packw(PRM(lb + 17), pw[l][6], 32, 32);
    packw(PRM(lb + 18), pw[l][7], 32, 32);
    pack_bm2_kernel<<<128, 256, 0, stream>>>(PRM(lb + 15), pbm[l][0]);
    pack_bm2_kernel<<<128, 256, 0, stream>>>(PRM(lb + 16), pbm[l][1]);
  }

  // --- two GNN layers ---
  for (int l = 0; l < 2; ++l) {
    int lb = l * 24;
    const float* eps_e = PRM(lb + 12);
    const float* eps_n = PRM(lb + 22);
    const float* efin1 = l ? ef1 : eA;
    const float* efin2 = l ? ef2 : eB;
    int din1 = l ? 32 : 64;
    int din2 = l ? 32 : 16;
    int kp1 = l ? 32 : 64;

    // edge convs: ef = LN(MLP2((1+eps_e)*ef))
    mlp_ln_kernel<<<E1_ / 64, 128, 0, stream>>>(
        efin1, nullptr, eps_e, pw[l][0], PRM(lb + 2), pw[l][1], PRM(lb + 3),
        PRM(lb + 5), PRM(lb + 4), ef1, din1, kp1);
    mlp_ln_kernel<<<E2_ / 64, 128, 0, stream>>>(
        efin2, nullptr, eps_e, pw[l][2], PRM(lb + 8), pw[l][3], PRM(lb + 9),
        PRM(lb + 11), PRM(lb + 10), ef2, din2, 32);

    // node conv aggregation
    hipMemsetAsync(conv, 0, (size_t)NA_ * 32 * 4, stream);
    nnconv_kernel<<<E1_ / 256, 128, 0, stream>>>(
        ef1, l ? hA : xA, src1, dst1, eps_n, pw[l][4], pbm[l][0], conv);
    nnconv_kernel<<<E2_ / 256, 128, 0, stream>>>(
        ef2, l ? hB : xB, src2, dst2, eps_n, pw[l][5], pbm[l][1], conv);

    // node updates: hA = LN(MLP2((1+eps_n)*hA + conv)); hB = LN(MLP2((1+eps_n)*hB))
    mlp_ln_kernel<<<NA_ / 64, 128, 0, stream>>>(
        l ? hA : xA, conv, eps_n, pw[l][6], PRM(lb + 19), pw[l][7], PRM(lb + 20),
        PRM(lb + 23), PRM(lb + 21), hA, 32, 32);
    mlp_ln_kernel<<<NB_ / 64, 128, 0, stream>>>(
        l ? hB : xB, nullptr, eps_n, pw[l][6], PRM(lb + 19), pw[l][7], PRM(lb + 20),
        PRM(lb + 23), PRM(lb + 21), hB, 32, 32);
  }

  // --- pooling + semantic attention + output head ---
  hipMemsetAsync(mA, 0, BGR_ * 32 * 4, stream);
  hipMemsetAsync(mB, 0, BGR_ * 32 * 4, stream);
  hipMemsetAsync(cA, 0, BGR_ * 4, stream);
  hipMemsetAsync(cB, 0, BGR_ * 4, stream);
  pool_reduce_kernel<<<(NA_ * 32 + 255) / 256, 256, 0, stream>>>(hA, gidA, NA_, mA, cA);
  pool_reduce_kernel<<<(NB_ * 32 + 255) / 256, 256, 0, stream>>>(hB, gidB, NB_, mB, cB);
  pool_out_kernel<<<1, 512, 0, stream>>>(
      mA, mB, cA, cB,
      (const float*)d_in[60], (const float*)d_in[62],  // P1, pb1
      (const float*)d_in[61], (const float*)d_in[63],  // P2, pb2
      (const float*)d_in[58], (const float*)d_in[59],  // Wo, bo
      (float*)d_out);
}
